// Quantize_77721728188640
// MI455X (gfx1250) — matrix operand order repaired and checked by reading, not compiled, not run on hardware
//
#include <hip/hip_runtime.h>
#include <hip/hip_bf16.h>

// ---------------------------------------------------------------------------
// VQ-VAE forward (eval) for x:[64,256,32,32] f32, embed:[1,1024,256] f32.
// Outputs (flat, concatenated): quantized [16777216], diff [16777216],
// indices [65536] (stored as float values), perplexity [1].
// Core: bf16 WMMA GEMM (x . e^T) with async global->LDS double-buffered
// codebook streaming; argmin of ||e||^2 - 2 x.e over K=1024.
// ---------------------------------------------------------------------------

#define K_CODES 1024
#define D_DIM   256
#define HW_SZ   1024           // H*W
#define M_TOTAL 65536          // B*H*W
#define M_TILE  64             // pixels per block
#define K_CHUNK 64             // codes staged per LDS chunk
#define N_KCHUNK (K_CODES / K_CHUNK)   // 16
#define E_ROW   264            // padded bf16 row (256 + 8) -> bank rotation
#define E_ROW_B (E_ROW * 2)    // 528 bytes

typedef __attribute__((ext_vector_type(16))) __bf16 v16bf;
typedef __attribute__((ext_vector_type(8)))  float  v8f;

union Frag { uint4 q[2]; v16bf v; };

__device__ __forceinline__ unsigned short f32_to_bf16(float f) {
  unsigned int u = __float_as_uint(f);
  unsigned int r = u + 0x7FFFu + ((u >> 16) & 1u);   // round-to-nearest-even
  return (unsigned short)(r >> 16);
}

// ---------------------------------------------------------------------------
// prep: e_sq[k] = sum_d embed[k][d]^2 ; bf16 copy of embed ; zero counts.
// grid = 1024 blocks, 256 threads.
// ---------------------------------------------------------------------------
__global__ __launch_bounds__(256) void vq_prep_kernel(
    const float* __restrict__ embed,
    unsigned short* __restrict__ embed_bf16,
    float* __restrict__ e_sq,
    unsigned int* __restrict__ counts) {
  const int k = blockIdx.x;
  const int t = threadIdx.x;
  const float v = embed[(size_t)k * D_DIM + t];
  embed_bf16[(size_t)k * D_DIM + t] = f32_to_bf16(v);

  float s = v * v;
  #pragma unroll
  for (int off = 16; off >= 1; off >>= 1) s += __shfl_xor(s, off);
  __shared__ float red[8];
  if ((t & 31) == 0) red[t >> 5] = s;
  __syncthreads();
  if (t == 0) {
    float tot = 0.f;
    #pragma unroll
    for (int i = 0; i < 8; ++i) tot += red[i];
    e_sq[k] = tot;
    counts[k] = 0u;   // re-zero every launch (graph replay safe)
  }
}

// ---------------------------------------------------------------------------
// main VQ kernel. grid = 1024 blocks x 256 threads (8 waves).
// Block handles 64 consecutive pixels (same image b, contiguous hw).
// Wave grid: 2 (m) x 4 (k); each wave owns TWO 16x16 m-subtiles per chunk
// (shared B fragment -> 2x arithmetic intensity, 16 wmma between barriers).
// Codebook chunks stream via global_load_async_to_lds_b128, double-buffered.
// ---------------------------------------------------------------------------
__global__ __launch_bounds__(256) void vq_main_kernel(
    const float* __restrict__ x,
    const float* __restrict__ embed_f32,
    const unsigned short* __restrict__ embed_bf16,
    const float* __restrict__ e_sq,
    unsigned int* __restrict__ counts,
    float* __restrict__ q_out,
    float* __restrict__ diff_out,
    float* __restrict__ ind_out) {

  __shared__ __align__(16) unsigned short lds_x[M_TILE][E_ROW];   // 33792 B
  __shared__ __align__(16) union SU {
    unsigned short e[2][K_CHUNK][E_ROW];                 // ping-pong, 67584 B
    struct { float s[M_TILE][65]; int k[M_TILE][65]; } cand;   // 33280 B
    float q[M_TILE][257];                                // gathered rows, 65792 B
  } u;
  __shared__ int best_idx[M_TILE];

  const int tid  = threadIdx.x;
  const int lane = tid & 31;
  const int wave = tid >> 5;
  const int wr   = wave >> 2;      // wave m-row: 0..1
  const int wc   = wave & 3;       // wave k-col: 0..3
  const int n    = lane & 15;      // tile column / A row
  const int hi   = lane >> 4;      // lane half

  const int m0  = blockIdx.x * M_TILE;
  const int bb  = m0 >> 10;        // image index (m0 / HW)
  const int hw0 = m0 & (HW_SZ - 1);
  const float* xbase = x + (((size_t)bb * D_DIM) << 10) + hw0;

  // LDS byte offsets of the two codebook buffers (flat-address low 32 bits
  // == wave-relative LDS offset on gfx1250).
  const unsigned lds_e[2] = {
    (unsigned)(size_t)(void*)&u.e[0][0][0],
    (unsigned)(size_t)(void*)&u.e[1][0][0] };

  // per-thread async-copy unit: 8 x b128 covering 64 rows x 512 B
  const int u_row = (tid >> 5);          // +8 per iteration
  const int u_c8  = tid & 31;
  const unsigned g_off0 = (unsigned)(u_row * 512 + u_c8 * 16);
  const unsigned l_off0 = (unsigned)(u_row * E_ROW_B + u_c8 * 16);

  // ---- prologue: kick off async copy of chunk 0 ---------------------------
  {
    const unsigned short* gb = embed_bf16;   // chunk 0 base
    #pragma unroll
    for (int it = 0; it < 8; ++it) {
      const unsigned go = g_off0 + (unsigned)(it * 8 * 512);
      const unsigned lo = lds_e[0] + l_off0 + (unsigned)(it * 8 * E_ROW_B);
      asm volatile("global_load_async_to_lds_b128 %0, %1, %2 offset:0"
                   :: "v"(lo), "v"(go), "s"(gb) : "memory");
    }
  }

  // ---- stage x tile: lds_x[p][d] = bf16(x[b, d, hw0+p]) -------------------
  {
    const int p = tid & 63;
    const int db = tid >> 6;
    #pragma unroll 4
    for (int it = 0; it < 64; ++it) {
      const int d = it * 4 + db;
      lds_x[p][d] = f32_to_bf16(xbase[((size_t)d << 10) + p]);
    }
  }

  float best0[8], best1[8];
  int   bk0[8], bk1[8];
  #pragma unroll
  for (int i = 0; i < 8; ++i) {
    best0[i] = 3.402823466e+38f; bk0[i] = 0;
    best1[i] = 3.402823466e+38f; bk1[i] = 0;
  }

  const v8f vzero = {0.f, 0.f, 0.f, 0.f, 0.f, 0.f, 0.f, 0.f};
  const unsigned short* a0row = &lds_x[wr * 16 + n][0];
  const unsigned short* a1row = &lds_x[32 + wr * 16 + n][0];

  for (int kc = 0; kc < N_KCHUNK; ++kc) {
    // ---- prefetch next chunk into spare buffer (async, ASYNCcnt) ---------
    if (kc + 1 < N_KCHUNK) {
      const unsigned short* gb = embed_bf16 + ((size_t)(kc + 1) * K_CHUNK << 8);
      const unsigned lbase = lds_e[(kc + 1) & 1] + l_off0;
      #pragma unroll
      for (int it = 0; it < 8; ++it) {
        const unsigned go = g_off0 + (unsigned)(it * 8 * 512);
        const unsigned lo = lbase + (unsigned)(it * 8 * E_ROW_B);
        asm volatile("global_load_async_to_lds_b128 %0, %1, %2 offset:0"
                     :: "v"(lo), "v"(go), "s"(gb) : "memory");
      }
      asm volatile("s_wait_asynccnt 0x8" ::: "memory");  // chunk kc landed
    } else {
      asm volatile("s_wait_asynccnt 0x0" ::: "memory");
    }
    __syncthreads();   // all waves see filled current buffer

    // ---- 2 x (16x16x256) dot tiles via 16 wmma, shared B fragment --------
    const unsigned short (*ebuf)[E_ROW] = u.e[kc & 1];
    const unsigned short* brow = &ebuf[wc * 16 + n][0];
    v8f acc0 = vzero, acc1 = vzero;
    #pragma unroll
    for (int dstep = 0; dstep < 8; ++dstep) {
      const int db = dstep * 32;
      Frag B, A0, A1;
      // B (32x16 bf16): col n; lanes<16 hold d {0..7,16..23}, lanes>=16 {8..15,24..31}
      B.q[0]  = *(const uint4*)&brow[db + hi * 8];
      B.q[1]  = *(const uint4*)&brow[db + 16 + hi * 8];
      // A (16x32 bf16): lanes<16 hold d {0..7,16..23}, lanes>=16 {8..15,24..31}
      A0.q[0] = *(const uint4*)&a0row[db + hi * 8];
      A0.q[1] = *(const uint4*)&a0row[db + 16 + hi * 8];
      A1.q[0] = *(const uint4*)&a1row[db + hi * 8];
      A1.q[1] = *(const uint4*)&a1row[db + 16 + hi * 8];
      acc0 = __builtin_amdgcn_wmma_f32_16x16x32_bf16(
                 false, A0.v, false, B.v, (short)0, acc0, false, false);
      acc1 = __builtin_amdgcn_wmma_f32_16x16x32_bf16(
                 false, A1.v, false, B.v, (short)0, acc1, false, false);
    }

    // ---- fold into running argmin: score = ||e||^2 - 2 x.e ---------------
    const int kg = kc * K_CHUNK + wc * 16 + n;   // same k across acc lanes
    const float esq = e_sq[kg];
    #pragma unroll
    for (int i = 0; i < 8; ++i) {
      const float s0 = fmaf(-2.0f, acc0[i], esq);
      if (s0 < best0[i]) { best0[i] = s0; bk0[i] = kg; }
      const float s1 = fmaf(-2.0f, acc1[i], esq);
      if (s1 < best1[i]) { best1[i] = s1; bk1[i] = kg; }
    }
    __syncthreads();   // everyone done with this buffer before refill
  }

  // ---- cross-wave argmin reduction ---------------------------------------
  #pragma unroll
  for (int i = 0; i < 8; ++i) {
    const int p = wr * 16 + hi * 8 + i;          // pixel (subtile 0)
    u.cand.s[p][wc * 16 + n]      = best0[i];
    u.cand.k[p][wc * 16 + n]      = bk0[i];
    u.cand.s[32 + p][wc * 16 + n] = best1[i];    // pixel (subtile 1)
    u.cand.k[32 + p][wc * 16 + n] = bk1[i];
  }
  __syncthreads();

  {
    const int p  = tid >> 2;       // pixel 0..63 (4 threads each)
    const int c0 = tid & 3;
    float bs = 3.402823466e+38f;
    int   bki = 0x7FFFFFFF;
    #pragma unroll
    for (int j = 0; j < 16; ++j) {
      const int c = c0 + j * 4;
      const float s = u.cand.s[p][c];
      const int   k2 = u.cand.k[p][c];
      if (s < bs || (s == bs && k2 < bki)) { bs = s; bki = k2; }
    }
    #pragma unroll
    for (int off = 2; off >= 1; off >>= 1) {     // xor stays inside 4-lane group
      const float os = __shfl_xor(bs, off);
      const int   ok = __shfl_xor(bki, off);
      if (os < bs || (os == bs && ok < bki)) { bs = os; bki = ok; }
    }
    if (c0 == 0) {
      best_idx[p] = bki;
      ind_out[m0 + p] = (float)bki;              // argmin index (exact in f32)
      atomicAdd(&counts[bki], 1u);
    }
  }
  __syncthreads();

  // ---- gather winning code rows into LDS (coalesced f32 rows) ------------
  for (int it = 0; it < M_TILE; ++it) {
    u.q[it][tid] = embed_f32[((size_t)best_idx[it] << 8) + tid];
  }
  __syncthreads();

  // ---- write quantized / diff, coalesced over 64 pixels per channel ------
  {
    const int p = tid & 63;
    const int dbse = tid >> 6;
    #pragma unroll 4
    for (int it = 0; it < 64; ++it) {
      const int d = it * 4 + dbse;
      const size_t off = (((size_t)bb * D_DIM + d) << 10) + hw0 + p;
      const float q  = u.q[p][d];
      const float xv = x[off];
      const float dd = q - xv;
      q_out[off]    = q;          // forward value of straight-through output
      diff_out[off] = dd * dd;
    }
  }
}

// ---------------------------------------------------------------------------
// perplexity from counts. 1 block, 256 threads.
// ---------------------------------------------------------------------------
__global__ __launch_bounds__(256) void vq_perp_kernel(
    const unsigned int* __restrict__ counts, float* __restrict__ out) {
  const int t = threadIdx.x;
  float s = 0.f;
  for (int k = t; k < K_CODES; k += 256) {
    const float p = (float)counts[k] * (1.0f / (float)M_TOTAL);
    s += p * __logf(p + 1e-10f);
  }
  #pragma unroll
  for (int off = 16; off >= 1; off >>= 1) s += __shfl_xor(s, off);
  __shared__ float red[8];
  if ((t & 31) == 0) red[t >> 5] = s;
  __syncthreads();
  if (t == 0) {
    float tot = 0.f;
    #pragma unroll
    for (int i = 0; i < 8; ++i) tot += red[i];
    *out = __expf(-tot);
  }
}

// ---------------------------------------------------------------------------
extern "C" void kernel_launch(void* const* d_in, const int* in_sizes, int n_in,
                              void* d_out, int out_size, void* d_ws, size_t ws_size,
                              hipStream_t stream) {
  const float* x     = (const float*)d_in[0];   // [64,256,32,32]
  const float* embed = (const float*)d_in[1];   // [1,1024,256]

  // workspace layout
  unsigned short* ebf16 = (unsigned short*)d_ws;                       // 512 KB
  float* esq            = (float*)((char*)d_ws + (size_t)K_CODES * D_DIM * 2);
  unsigned int* counts  = (unsigned int*)((char*)esq + K_CODES * sizeof(float));

  // output layout (floats)
  float* q_out    = (float*)d_out;
  float* diff_out = q_out + (size_t)16777216;
  float* ind_out  = q_out + (size_t)33554432;
  float* perp_out = q_out + (size_t)33554432 + 65536;

  vq_prep_kernel<<<K_CODES, 256, 0, stream>>>(embed, ebf16, esq, counts);
  vq_main_kernel<<<M_TOTAL / M_TILE, 256, 0, stream>>>(
      x, embed, ebf16, esq, counts, q_out, diff_out, ind_out);
  vq_perp_kernel<<<1, 256, 0, stream>>>(counts, perp_out);
}
